// AttentionHead_67869073211663
// MI455X (gfx1250) — matrix operand order repaired and checked by reading, not compiled, not run on hardware
//
#include <hip/hip_runtime.h>
#include <hip/hip_bf16.h>

typedef unsigned int   u32;
typedef unsigned short u16;

typedef __attribute__((ext_vector_type(16))) __bf16 v16bf;
typedef __attribute__((ext_vector_type(8)))  float  v8f;
typedef __attribute__((ext_vector_type(8)))  u32    v8u;
typedef __attribute__((ext_vector_type(4)))  u32    v4u;
typedef __attribute__((ext_vector_type(8)))  int    v8i;
typedef __attribute__((ext_vector_type(4)))  int    v4i;

// ---------------------------------------------------------------------------
// fp32 -> bf16 (round to nearest even)
// ---------------------------------------------------------------------------
__device__ __forceinline__ u16 f2bf(float x) {
    u32 u = __builtin_bit_cast(u32, x);
    u32 r = u + 0x7FFFu + ((u >> 16) & 1u);
    return (u16)(r >> 16);
}

// ---------------------------------------------------------------------------
// WMMA fragment loaders, wave32, per CDNA5 ISA 7.12.2 layouts.
// A fragment 16x32 bf16 (M x K): lane l (0-15) row=l, K {hi*8..+7, 16+hi*8..+7}
// B fragment 32x16 bf16 (K x N), stored [N][K] row-major:
//   lane l col=l&15, K {hi*8..+7, 16+hi*8..+7} (same as A)
// ---------------------------------------------------------------------------
__device__ __forceinline__ v16bf load_a_frag(const u16* A0, int lda, int lane) {
    int l = lane & 15, hi = lane >> 4;
    const u32* p0 = (const u32*)(A0 + (size_t)l * lda + hi * 8);
    const u32* p1 = (const u32*)(A0 + (size_t)l * lda + 16 + hi * 8);
    v8u u = { p0[0], p0[1], p0[2], p0[3], p1[0], p1[1], p1[2], p1[3] };
    return __builtin_bit_cast(v16bf, u);
}

__device__ __forceinline__ v16bf load_b_frag(const u16* B0, int ldb, int lane) {
    int n = lane & 15, koff = (lane >> 4) * 8;
    const u32* p = (const u32*)(B0 + (size_t)n * ldb + koff);
    v8u u = { p[0], p[1], p[2], p[3], p[8], p[9], p[10], p[11] };
    return __builtin_bit_cast(v16bf, u);
}

__device__ __forceinline__ v8f wmma_bf16(v16bf a, v16bf b, v8f c) {
    return __builtin_amdgcn_wmma_f32_16x16x32_bf16(
        /*neg_a=*/false, a, /*neg_b=*/false, b,
        /*c_mod=*/(short)0, c, /*reuse_a=*/false, /*reuse_b=*/false);
}

// ---------------------------------------------------------------------------
// Tensor Data Mover: DMA a 2D bf16 tile [64 rows x 32 K-elems] from global
// (row stride = stride_elems) into LDS at byte offset lds_off.
// D# built per CDNA5 ISA (groups 0/1; 2D -> rest zero).
// This toolchain exposes the 6-arg builtin (clang-23 / therock-10.0 headers).
// ---------------------------------------------------------------------------
__device__ __forceinline__ void tdm_load_b_tile(u32 lds_off, const u16* gptr,
                                                u32 stride_elems) {
    unsigned long long ga = (unsigned long long)(uintptr_t)gptr;
    const u32 tile_k = 32, tile_n = 64;
    v4u g0 = { 1u,                                   // count=1, user descriptor
               lds_off,                              // lds_addr[31:0]
               (u32)(ga & 0xFFFFFFFFu),              // global_addr[31:0]
               (u32)((ga >> 32) & 0x01FFFFFFu) | 0x80000000u }; // addr[56:32] | type=2
    v8i g1 = { (int)(1u << 16),                      // data_size=1 (2B), mask=0
               (int)((stride_elems & 0xFFFFu) << 16),          // tensor_dim0[15:0]
               (int)(((stride_elems >> 16) & 0xFFFFu) | 0xFFFF0000u), // td0 hi | tensor_dim1=0xFFFF
               (int)(tile_k << 16),                  // tensor_dim1 hi=0 | tile_dim0
               (int)tile_n,                          // tile_dim1 | tile_dim2=0
               (int)stride_elems,                    // tensor_dim0_stride[31:0]
               0,                                    // stride hi | dim1_stride lo
               0 };
    v4i z4 = { 0, 0, 0, 0 };
    v8i z8 = { 0, 0, 0, 0, 0, 0, 0, 0 };
    __builtin_amdgcn_tensor_load_to_lds(g0, g1, z4, z4, z8, 0);
}

// ---------------------------------------------------------------------------
// f32 -> bf16 convert
// ---------------------------------------------------------------------------
__global__ void k_cvt(const float* __restrict__ s, u16* __restrict__ d, int n) {
    int i = blockIdx.x * 256 + threadIdx.x;
    if (i < n) d[i] = f2bf(s[i]);
}

// ---------------------------------------------------------------------------
// GEMM: C[M,N] = A[M,768] * B[N,768]^T  (bf16 in, f32 accumulate)
// Block = 4 waves = 128(M) x 64(N); one wave = 32(M) x 64(N).
// B k-chunks staged through LDS by the Tensor Data Mover (double-buffered),
// shared by all 4 waves.
// mode 0: Q -> q_bh [BH][T][D] bf16 (scaled 1/8)   mode 1: K -> k_bh
// mode 2: V -> v_t  [BH][D][T] bf16 (transposed)   mode 3: out f32 [M][768]
// ---------------------------------------------------------------------------
__global__ __launch_bounds__(128)
void k_gemm(const u16* __restrict__ A, const u16* __restrict__ Bw,
            u16* __restrict__ dst16, float* __restrict__ dst32,
            int mode, float scale) {
    __shared__ u16 bstage[2][64 * 32];     // double-buffered B tile, 2 x 4 KB

    int tid  = threadIdx.x;
    int wave = tid >> 5, lane = tid & 31;
    int l = lane & 15, hi = lane >> 4;
    int m0 = blockIdx.y * 128 + wave * 32;
    int n0 = blockIdx.x * 64;

    const int NI = 768 / 32;

    if (wave == 0)
        tdm_load_b_tile((u32)(uintptr_t)&bstage[0][0],
                        Bw + (size_t)n0 * 768, 768);

    v8f acc[2][4] = {};
    for (int i = 0; i < NI; ++i) {
        int kc = i * 32;
        if (wave == 0) {
            __builtin_amdgcn_s_wait_tensorcnt(0);      // buf[i&1] landed
        }
        __syncthreads();                               // publish tile to block
        if (wave == 0 && i + 1 < NI)                   // prefetch next chunk
            tdm_load_b_tile((u32)(uintptr_t)&bstage[(i + 1) & 1][0],
                            Bw + (size_t)n0 * 768 + (kc + 32), 768);

        v16bf a0 = load_a_frag(A + (size_t)m0 * 768 + kc,        768, lane);
        v16bf a1 = load_a_frag(A + (size_t)(m0 + 16) * 768 + kc, 768, lane);
        const u16* bl = &bstage[i & 1][0];
#pragma unroll
        for (int j = 0; j < 4; ++j) {
            v16bf b = load_b_frag(bl + (size_t)(16 * j) * 32, 32, lane);
            acc[0][j] = wmma_bf16(a0, b, acc[0][j]);
            acc[1][j] = wmma_bf16(a1, b, acc[1][j]);
        }
        __syncthreads();                               // buf reuse fence
    }

#pragma unroll
    for (int mi = 0; mi < 2; ++mi) {
#pragma unroll
        for (int j = 0; j < 4; ++j) {
#pragma unroll
            for (int r = 0; r < 8; ++r) {
                int gm = m0 + mi * 16 + r + hi * 8;    // C layout rows
                int gn = n0 + j * 16 + l;
                float v = acc[mi][j][r] * scale;
                if (mode == 3) {
                    dst32[(size_t)gm * 768 + gn] = v;
                } else {
                    int bh = (gm >> 11) * 12 + (gn >> 6);
                    int t  = gm & 2047;
                    int d  = gn & 63;
                    if (mode == 2)
                        dst16[((size_t)bh * 64 + d) * 2048 + t] = f2bf(v);
                    else
                        dst16[((size_t)bh * 2048 + t) * 64 + d] = f2bf(v);
                }
            }
        }
    }
}

// ---------------------------------------------------------------------------
// Flash attention, causal. One wave = 16 query rows; 8 waves/block.
// S = Q K^T (Q pre-scaled), online softmax in f32, O += P V (V transposed).
// ---------------------------------------------------------------------------
__global__ __launch_bounds__(256)
void k_attn(const u16* __restrict__ q, const u16* __restrict__ kk,
            const u16* __restrict__ vt, u16* __restrict__ att) {
    __shared__ u16 lds_p[8][16 * 32];       // per-wave P-tile transpose buffer

    int tid  = threadIdx.x;
    int wave = tid >> 5, lane = tid & 31;
    int l = lane & 15, hi = lane >> 4;

    int bh = blockIdx.y;
    int b  = bh / 12, h = bh % 12;
    int m0 = (blockIdx.x * 8 + wave) * 16;

    const u16* qh = q  + (size_t)bh * 2048 * 64;
    const u16* kh = kk + (size_t)bh * 2048 * 64;
    const u16* vh = vt + (size_t)bh * 64 * 2048;   // [d][t]

    v16bf aq0 = load_a_frag(qh + (size_t)m0 * 64,      64, lane);
    v16bf aq1 = load_a_frag(qh + (size_t)m0 * 64 + 32, 64, lane);

    float run_max[8], run_sum[8];
#pragma unroll
    for (int r = 0; r < 8; ++r) { run_max[r] = -3.0e38f; run_sum[r] = 0.0f; }
    v8f acc[4] = {};

    int row_base = m0 + hi * 8;

    for (int s0 = 0; s0 <= m0 + 15; s0 += 32) {
        v16bf b00 = load_b_frag(kh + (size_t)s0 * 64,             64, lane);
        v16bf b01 = load_b_frag(kh + (size_t)s0 * 64 + 32,        64, lane);
        v16bf b10 = load_b_frag(kh + (size_t)(s0 + 16) * 64,      64, lane);
        v16bf b11 = load_b_frag(kh + (size_t)(s0 + 16) * 64 + 32, 64, lane);
        v8f z = {};
        v8f sc0 = wmma_bf16(aq0, b00, z);  sc0 = wmma_bf16(aq1, b01, sc0);
        v8f sc1 = wmma_bf16(aq0, b10, z);  sc1 = wmma_bf16(aq1, b11, sc1);

        int c0 = s0 + l, c1 = c0 + 16;
#pragma unroll
        for (int r = 0; r < 8; ++r) {
            int row = row_base + r;
            float a  = (c0 <= row) ? sc0[r] : -1.0e30f;   // causal mask
            float bb = (c1 <= row) ? sc1[r] : -1.0e30f;
            float t = fmaxf(a, bb);
            t = fmaxf(t, __shfl_xor(t, 1, 32));
            t = fmaxf(t, __shfl_xor(t, 2, 32));
            t = fmaxf(t, __shfl_xor(t, 4, 32));
            t = fmaxf(t, __shfl_xor(t, 8, 32));
            float nm = fmaxf(run_max[r], t);
            float rs_scale = __expf(run_max[r] - nm);
            run_max[r] = nm;
            float pa = __expf(a - nm);
            float pb = __expf(bb - nm);
            float rs = pa + pb;
            rs += __shfl_xor(rs, 1, 32);
            rs += __shfl_xor(rs, 2, 32);
            rs += __shfl_xor(rs, 4, 32);
            rs += __shfl_xor(rs, 8, 32);
            run_sum[r] = run_sum[r] * rs_scale + rs;
#pragma unroll
            for (int j = 0; j < 4; ++j) acc[j][r] *= rs_scale;
            sc0[r] = pa;
            sc1[r] = pb;
        }

        // transpose P (C layout -> A layout) through per-wave LDS
#pragma unroll
        for (int r = 0; r < 8; ++r) {
            lds_p[wave][(r + hi * 8) * 32 + l]      = f2bf(sc0[r]);
            lds_p[wave][(r + hi * 8) * 32 + 16 + l] = f2bf(sc1[r]);
        }
        asm volatile("s_wait_dscnt 0" ::: "memory");
        v16bf ap = load_a_frag(&lds_p[wave][0], 32, lane);

#pragma unroll
        for (int j = 0; j < 4; ++j) {
            v16bf bv = load_b_frag(vh + (size_t)(16 * j) * 2048 + s0, 2048, lane);
            acc[j] = wmma_bf16(ap, bv, acc[j]);
        }
    }

#pragma unroll
    for (int j = 0; j < 4; ++j) {
#pragma unroll
        for (int r = 0; r < 8; ++r) {
            int t = m0 + r + hi * 8;
            int c = h * 64 + j * 16 + l;
            float v = acc[j][r] * (1.0f / run_sum[r]);
            att[((size_t)(b * 2048 + t)) * 768 + c] = f2bf(v);
        }
    }
}

// ---------------------------------------------------------------------------
// launcher
// ---------------------------------------------------------------------------
extern "C" void kernel_launch(void* const* d_in, const int* in_sizes, int n_in,
                              void* d_out, int out_size, void* d_ws, size_t ws_size,
                              hipStream_t stream) {
    const float* x  = (const float*)d_in[0];
    const float* Wq = (const float*)d_in[1];
    const float* Wk = (const float*)d_in[2];
    const float* Wv = (const float*)d_in[3];
    const float* Wp = (const float*)d_in[4];

    const int XB = 8192 * 768;   // 6,291,456
    const int WE = 768 * 768;    //   589,824

    char* p = (char*)d_ws;
    u16* xb  = (u16*)p; p += (size_t)XB * 2;   // reused as 'att' after QKV
    u16* wqb = (u16*)p; p += (size_t)WE * 2;
    u16* wkb = (u16*)p; p += (size_t)WE * 2;
    u16* wvb = (u16*)p; p += (size_t)WE * 2;
    u16* wpb = (u16*)p; p += (size_t)WE * 2;
    u16* qb  = (u16*)p; p += (size_t)XB * 2;
    u16* kb  = (u16*)p; p += (size_t)XB * 2;
    u16* vtb = (u16*)p; p += (size_t)XB * 2;
    u16* att = xb;

    k_cvt<<<XB / 256, 256, 0, stream>>>(x,  xb,  XB);
    k_cvt<<<WE / 256, 256, 0, stream>>>(Wq, wqb, WE);
    k_cvt<<<WE / 256, 256, 0, stream>>>(Wk, wkb, WE);
    k_cvt<<<WE / 256, 256, 0, stream>>>(Wv, wvb, WE);
    k_cvt<<<WE / 256, 256, 0, stream>>>(Wp, wpb, WE);

    dim3 ggrid(12, 64);   // N/64 x M/128
    k_gemm<<<ggrid, 128, 0, stream>>>(xb, wqb, qb,  nullptr, 0, 0.125f);
    k_gemm<<<ggrid, 128, 0, stream>>>(xb, wkb, kb,  nullptr, 1, 1.0f);
    k_gemm<<<ggrid, 128, 0, stream>>>(xb, wvb, vtb, nullptr, 2, 1.0f);

    k_attn<<<dim3(16, 48), 256, 0, stream>>>(qb, kb, vtb, att);

    k_gemm<<<ggrid, 128, 0, stream>>>(att, wpb, nullptr, (float*)d_out, 3, 1.0f);
}
